// PointAdapter_64261300683152
// MI455X (gfx1250) — matrix operand order repaired and checked by reading, not compiled, not run on hardware
//
#include <hip/hip_runtime.h>
#include <hip/hip_bf16.h>
#include <cstdint>
#include <cstddef>

// Problem constants (reference: B=2,F=16,N=128,C=1280,MID=128,h=w=512 -> H0=W0=32)
#define BFN   32            // B*F
#define NPTS  128
#define CIN   1280
#define MIDC  128
#define MROWS 32768         // BFN*32*32

typedef __bf16 bf16;
typedef __bf16 v16bf __attribute__((ext_vector_type(16)));
typedef __bf16 v8bf  __attribute__((ext_vector_type(8)));
typedef float  v8f   __attribute__((ext_vector_type(8)));
typedef float  v4f   __attribute__((ext_vector_type(4)));

union FragU { v8bf h[2]; v16bf v; };

// A fragment (16x32 bf16, row-major [M,K]): lane (half,lm) -> row lm,
// K chunks {kb+half*8 .. +7} and {+16 .. +23} (per CDNA5 16-bit A layout).
__device__ __forceinline__ v16bf loadA(const bf16* __restrict__ base, int row, int ldk,
                                       int kb, int half) {
  const bf16* p = base + (size_t)row * ldk + kb + half * 8;
  FragU u;
  u.h[0] = *(const v8bf*)(p);
  u.h[1] = *(const v8bf*)(p + 16);
  return u.v;
}

// B fragment (32x16 bf16) from transposed weights [N,K]: lane holds column n,
// K chunks {kb+half*8 .. +7} and {+16 .. +23} (same order as A).
__device__ __forceinline__ v16bf loadB(const bf16* __restrict__ base, int col, int ldk,
                                       int kb, int half) {
  const bf16* p = base + (size_t)col * ldk + kb + half * 8;
  FragU u;
  u.h[0] = *(const v8bf*)(p);
  u.h[1] = *(const v8bf*)(p + 16);
  return u.v;
}

__device__ __forceinline__ v8f wmma_bf16(v16bf a, v16bf b, v8f c) {
  return __builtin_amdgcn_wmma_f32_16x16x32_bf16(false, a, false, b, (short)0, c,
                                                 false, false);
}

// ---------------- scatter: mean-pool point embeddings into 32x32 grid ----------------
__global__ void scatter_kernel(const float* __restrict__ pe, const float* __restrict__ kp,
                               const int* __restrict__ sub, float* __restrict__ sums,
                               float* __restrict__ hits) {
  int p = blockIdx.x;                 // 0 .. BFN*NPTS-1
  int bf = p / NPTS;
  int b  = bf >> 4;                   // F = 16
  int n  = p - bf * NPTS;
  float kx = kp[p * 2 + 0], ky = kp[p * 2 + 1];
  if (!((fminf(kx, ky) >= 0.f) && (sub[p] != -1))) return;
  int px = min(max((int)floorf(kx * 32.f), 0), 31);
  int py = min(max((int)floorf(ky * 32.f), 0), 31);
  int cell = bf * 1024 + py * 32 + px;
  const float* e = pe + ((size_t)b * NPTS + n) * CIN;
  float* s = sums + (size_t)cell * CIN;
  for (int c = threadIdx.x; c < CIN; c += blockDim.x) atomicAdd(&s[c], e[c]);
  if (threadIdx.x == 0) atomicAdd(&hits[cell], 1.f);
}

// -------- normalize: grid = sums/max(hits,1), split into bf16 hi/lo; build mask --------
__global__ void normalize_kernel(const float* __restrict__ sums, const float* __restrict__ hits,
                                 bf16* __restrict__ ghi, bf16* __restrict__ glo,
                                 float* __restrict__ mask) {
  size_t i = (size_t)blockIdx.x * blockDim.x + threadIdx.x;
  if (i >= (size_t)MROWS * CIN) return;
  int m = (int)(i / CIN);
  float h = hits[m];
  float g = sums[i] / fmaxf(h, 1.f);
  bf16 hi = (bf16)g;
  ghi[i] = hi;
  glo[i] = (bf16)(g - (float)hi);
  if ((i % CIN) == 0) mask[m] = (h > 0.f) ? 1.f : 0.f;
}

// ------------- weight prep: transpose to [N,K] and split into bf16 hi/lo -------------
__global__ void convw1_kernel(const float* __restrict__ w1s, bf16* __restrict__ hi,
                              bf16* __restrict__ lo) {
  size_t i = (size_t)blockIdx.x * blockDim.x + threadIdx.x;   // 4*128*1280
  if (i >= (size_t)4 * MIDC * CIN) return;
  int k = (int)(i % CIN);
  int n = (int)((i / CIN) % MIDC);
  int l = (int)(i / ((size_t)CIN * MIDC));
  float v = w1s[((size_t)l * CIN + k) * MIDC + n];
  bf16 h = (bf16)v;
  hi[i] = h;
  lo[i] = (bf16)(v - (float)h);
}

__global__ void convw2_kernel(const float* __restrict__ w2, bf16* __restrict__ hi,
                              bf16* __restrict__ lo, int ch) {
  size_t i = (size_t)blockIdx.x * blockDim.x + threadIdx.x;   // ch*128
  if (i >= (size_t)ch * MIDC) return;
  int k = (int)(i % MIDC);
  int n = (int)(i / MIDC);
  float v = w2[(size_t)k * ch + n];
  bf16 h = (bf16)v;
  hi[i] = h;
  lo[i] = (bf16)(v - (float)h);
}

// ------------------- GEMM1: hid = silu(grid @ w1 + b1), split-bf16 -------------------
__device__ __forceinline__ void store_hid(v8f acc, int mb, int n,
                                          const float* __restrict__ b1,
                                          bf16* __restrict__ Hhi, bf16* __restrict__ Hlo) {
  float bias = b1[n];
#pragma unroll
  for (int r = 0; r < 8; ++r) {
    float v = acc[r] + bias;
    float s = v * (1.f / (1.f + __expf(-v)));     // silu
    bf16 hi = (bf16)s;
    size_t idx = (size_t)(mb + r) * MIDC + n;
    Hhi[idx] = hi;
    Hlo[idx] = (bf16)(s - (float)hi);
  }
}

__global__ __launch_bounds__(256) void gemm1_kernel(
    const bf16* __restrict__ Ahi, const bf16* __restrict__ Alo,
    const bf16* __restrict__ Whi, const bf16* __restrict__ Wlo,   // [128,1280] (N,K)
    const float* __restrict__ b1, bf16* __restrict__ Hhi, bf16* __restrict__ Hlo) {
  const int wave = threadIdx.x >> 5, lane = threadIdx.x & 31;
  const int half = lane >> 4, lm = lane & 15;
  const int m0 = blockIdx.x * 32 + (wave >> 2) * 16;
  const int n0 = (wave & 3) * 16;          // second tile at n0+64
  const int rowA = m0 + lm;
  v8f acc0 = {}, acc1 = {};
  for (int kb = 0; kb < CIN; kb += 32) {
    v16bf ah  = loadA(Ahi, rowA, CIN, kb, half);
    v16bf al  = loadA(Alo, rowA, CIN, kb, half);
    v16bf b0h = loadB(Whi, n0 + lm, CIN, kb, half);
    v16bf b0l = loadB(Wlo, n0 + lm, CIN, kb, half);
    v16bf b1h = loadB(Whi, n0 + 64 + lm, CIN, kb, half);
    v16bf b1l = loadB(Wlo, n0 + 64 + lm, CIN, kb, half);
    acc0 = wmma_bf16(ah, b0h, acc0);
    acc0 = wmma_bf16(ah, b0l, acc0);
    acc0 = wmma_bf16(al, b0h, acc0);
    acc1 = wmma_bf16(ah, b1h, acc1);
    acc1 = wmma_bf16(ah, b1l, acc1);
    acc1 = wmma_bf16(al, b1h, acc1);
  }
  const int mb = m0 + half * 8;
  store_hid(acc0, mb, n0 + lm, b1, Hhi, Hlo);
  store_hid(acc1, mb, n0 + 64 + lm, b1, Hhi, Hlo);
}

// --------- GEMM2: feat = (hid @ w2 + b2) * mask; level0 -> transposed output ---------
__global__ __launch_bounds__(256) void gemm2_kernel(
    const bf16* __restrict__ Ahi, const bf16* __restrict__ Alo,   // hid [M,128]
    const bf16* __restrict__ Whi, const bf16* __restrict__ Wlo,   // w2t [ch,128]
    const float* __restrict__ b2, const float* __restrict__ mask,
    int ch, int level0, float* __restrict__ dst) {
  const int wave = threadIdx.x >> 5, lane = threadIdx.x & 31;
  const int half = lane >> 4, lm = lane & 15;
  const int m0 = blockIdx.x * 32 + (wave >> 2) * 16;
  const int n0 = blockIdx.y * 64 + (wave & 3) * 16;
  const int rowA = m0 + lm, colB = n0 + lm;
  v8f acc = {};
#pragma unroll
  for (int kb = 0; kb < MIDC; kb += 32) {
    v16bf ah = loadA(Ahi, rowA, MIDC, kb, half);
    v16bf al = loadA(Alo, rowA, MIDC, kb, half);
    v16bf bh = loadB(Whi, colB, MIDC, kb, half);
    v16bf bl = loadB(Wlo, colB, MIDC, kb, half);
    acc = wmma_bf16(ah, bh, acc);
    acc = wmma_bf16(ah, bl, acc);
    acc = wmma_bf16(al, bh, acc);
  }
  const int mb = m0 + half * 8;
  v4f mk0 = *(const v4f*)(mask + mb);
  v4f mk1 = *(const v4f*)(mask + mb + 4);
  float bias = b2[colB];
  float f[8];
#pragma unroll
  for (int r = 0; r < 8; ++r)
    f[r] = (acc[r] + bias) * ((r < 4) ? mk0[r] : mk1[r - 4]);
  if (level0) {
    // out[bf, n, pix]: 8 consecutive pixels per lane -> two b128 stores
    const int bfi = mb >> 10, pix = mb & 1023;
    float* o = dst + ((size_t)bfi * ch + colB) * 1024 + pix;
    v4f s0 = {f[0], f[1], f[2], f[3]};
    v4f s1 = {f[4], f[5], f[6], f[7]};
    *(v4f*)o = s0;
    *(v4f*)(o + 4) = s1;
  } else {
#pragma unroll
    for (int r = 0; r < 8; ++r) dst[(size_t)(mb + r) * ch + colB] = f[r];
  }
}

// ------- fused bilinear (half-pixel, exact /2 or /4) + nearest-mask + transpose -------
__global__ void resize_kernel(const float* __restrict__ feat, const float* __restrict__ mask,
                              float* __restrict__ out, int ch, int h2, int step, int soff) {
  size_t i = (size_t)blockIdx.x * blockDim.x + threadIdx.x;
  size_t total = (size_t)BFN * ch * h2 * h2;
  if (i >= total) return;
  int x2 = (int)(i % h2);
  int y2 = (int)((i / h2) % h2);
  int n  = (int)((i / ((size_t)h2 * h2)) % ch);
  int bf = (int)(i / ((size_t)h2 * h2 * ch));
  int ys = y2 * step + soff, xs = x2 * step + soff;
  size_t base = (size_t)bf * 1024;
  float s = feat[(base + ys * 32 + xs) * ch + n] +
            feat[(base + ys * 32 + xs + 1) * ch + n] +
            feat[(base + (ys + 1) * 32 + xs) * ch + n] +
            feat[(base + (ys + 1) * 32 + xs + 1) * ch + n];
  float m = mask[base + (y2 * step) * 32 + x2 * step];
  out[i] = 0.25f * s * m;
}

__global__ void fill_ones_kernel(float* __restrict__ p, int n) {
  int i = blockIdx.x * blockDim.x + threadIdx.x;
  if (i < n) p[i] = 1.f;
}

extern "C" void kernel_launch(void* const* d_in, const int* in_sizes, int n_in,
                              void* d_out, int out_size, void* d_ws, size_t ws_size,
                              hipStream_t stream) {
  (void)in_sizes; (void)n_in; (void)out_size; (void)ws_size;
  const float* pe  = (const float*)d_in[0];
  const float* kp  = (const float*)d_in[1];
  const int*   sub = (const int*)d_in[2];
  // d_in[3] = pose_latents: unused by the reference computation
  const float* w1s = (const float*)d_in[4];
  const float* b1s = (const float*)d_in[5];
  const float* w2p[4] = {(const float*)d_in[6], (const float*)d_in[8],
                         (const float*)d_in[10], (const float*)d_in[12]};
  const float* b2p[4] = {(const float*)d_in[7], (const float*)d_in[9],
                         (const float*)d_in[11], (const float*)d_in[13]};

  // workspace layout (4KB-aligned offsets); total ~357 MB; sums region reused as feat
  char* ws = (char*)d_ws;
  float* sums  = (float*)(ws + 0);              // [32768,1280] f32 (later: feat)
  float* feat  = sums;
  bf16*  ghi   = (bf16*)(ws + 167772160);       // grid hi  [32768,1280]
  bf16*  glo   = (bf16*)(ws + 251658240);       // grid lo
  bf16*  hhi   = (bf16*)(ws + 335544320);       // hid hi   [32768,128]
  bf16*  hlo   = (bf16*)(ws + 343932928);       // hid lo
  bf16*  w1thi = (bf16*)(ws + 352321536);       // [4][128,1280]
  bf16*  w1tlo = (bf16*)(ws + 353632256);
  bf16*  w2thi = (bf16*)(ws + 354942976);       // [sum(ch),128]
  bf16*  w2tlo = (bf16*)(ws + 355844096);
  float* hits  = (float*)(ws + 356745216);      // [32768]
  float* mask  = (float*)(ws + 356876288);      // [32768]

  float* out = (float*)d_out;
  float* out_l[4] = {out, out + 10485760, out + 15728640, out + 18350080};
  float* lossm = out + 20971520;

  hipMemsetAsync(sums, 0, (size_t)MROWS * CIN * sizeof(float), stream);
  hipMemsetAsync(hits, 0, (size_t)MROWS * sizeof(float), stream);

  scatter_kernel<<<BFN * NPTS, 256, 0, stream>>>(pe, kp, sub, sums, hits);
  {
    size_t tot = (size_t)MROWS * CIN;
    normalize_kernel<<<(unsigned)((tot + 255) / 256), 256, 0, stream>>>(sums, hits, ghi, glo, mask);
  }
  convw1_kernel<<<(4 * MIDC * CIN + 255) / 256, 256, 0, stream>>>(w1s, w1thi, w1tlo);

  const int CH[4]    = {320, 640, 1280, 1280};
  const int w2off[4] = {0, 320 * 128, 960 * 128, 2240 * 128};
  for (int l = 0; l < 4; ++l)
    convw2_kernel<<<(CH[l] * MIDC + 255) / 256, 256, 0, stream>>>(
        w2p[l], w2thi + w2off[l], w2tlo + w2off[l], CH[l]);

  const int h2s[4]  = {32, 16, 8, 8};
  const int step[4] = {1, 2, 4, 4};
  const int soff[4] = {0, 0, 1, 1};
  for (int l = 0; l < 4; ++l) {
    gemm1_kernel<<<MROWS / 32, 256, 0, stream>>>(
        ghi, glo, w1thi + (size_t)l * MIDC * CIN, w1tlo + (size_t)l * MIDC * CIN,
        b1s + l * MIDC, hhi, hlo);
    float* dst = (l == 0) ? out_l[0] : feat;
    gemm2_kernel<<<dim3(MROWS / 32, CH[l] / 64), 256, 0, stream>>>(
        hhi, hlo, w2thi + w2off[l], w2tlo + w2off[l], b2p[l], mask, CH[l],
        (l == 0) ? 1 : 0, dst);
    if (l > 0) {
      size_t tot = (size_t)BFN * CH[l] * h2s[l] * h2s[l];
      resize_kernel<<<(unsigned)((tot + 255) / 256), 256, 0, stream>>>(
          feat, mask, out_l[l], CH[l], h2s[l], step[l], soff[l]);
    }
  }
  fill_ones_kernel<<<512, 256, 0, stream>>>(lossm, 131072);
}
